// WaveNetModel_6210522710160
// MI455X (gfx1250) — matrix operand order repaired and checked by reading, not compiled, not run on hardware
//
#include <hip/hip_runtime.h>

// ---------------------------------------------------------------------------
// Fused WaveNet forward for MI455X (gfx1250, wave32, WMMA bf16).
//
// Roofline: ~47 GFLOP vs ~300MB HBM traffic (h=268MB dominates) -> bandwidth
// bound (~15-30us @ 23.3TB/s). Design: ONE pass over h; residual stream +
// skip accumulator stay on-chip (LDS + VGPRs). bf16 v_wmma_f32_16x16x32_bf16
// for all matmuls, f32 accumulation.
// ---------------------------------------------------------------------------

typedef __attribute__((ext_vector_type(16))) __bf16 v16bf;
typedef __attribute__((ext_vector_type(8)))  float  v8f;

union BfTile { v16bf v; uint4 q[2]; };

#define Tlen  8192
#define NBk   16
#define TT    128     // output samples per workgroup
#define HALO  512     // >= sum of dilations (510), chunk aligned
#define NC    (TT + HALO)   // 640 buffer columns

// bf16 weight arena offsets in d_ws (elements)
#define OFF_EMB    0          // [256][64]
#define OFF_WCONV  16384      // [16][2 taps][128][64]
#define OFF_WRES   278528     // [16][64][64]
#define OFF_WSKIP  344064     // [16][256][64]
#define OFF_WOUT1  606208     // [256][256]
#define OFF_WOUT2  671744     // [256][256]
#define WS_ELEMS   737280

__device__ constexpr int kDil[NBk] = {1,2,4,8,16,32,64,128,1,2,4,8,16,32,64,128};
// first 128-aligned chunk each block must compute (floor of 512 - suffix-dil-sum)
__device__ constexpr int kCh0[NBk] = {0,0,0,0,0,0,128,256,256,256,256,256,256,256,384,512};

static __device__ inline v8f vzero() {
  v8f v;
#pragma unroll
  for (int i = 0; i < 8; ++i) v[i] = 0.f;
  return v;
}

static __device__ inline unsigned pk2(float a, float b) {   // 2xf32 -> packed bf16 (RNE)
  unsigned ia = __float_as_uint(a), ib = __float_as_uint(b);
  ia = (ia + 0x7FFFu + ((ia >> 16) & 1u)) >> 16;
  ib = (ib + 0x7FFFu + ((ib >> 16) & 1u)) >> 16;
  return (ia & 0xFFFFu) | (ib << 16);
}
static __device__ inline float b2f(unsigned u) { return __uint_as_float(u << 16); }

static __device__ inline v8f wmma_bf(v16bf a, v16bf b, v8f c) {
  return __builtin_amdgcn_wmma_f32_16x16x32_bf16(false, a, false, b, (short)0, c,
                                                 false, false);
}

// A tile 16x32 bf16 from row-major [M][ldk] memory.
// lane<16: rows m0+lane, K {k0..k0+7, k0+16..k0+23}; lanes>=16: +8 on both K chunks.
static __device__ inline v16bf load_a(const unsigned short* w, int m0, int k0,
                                      int ldk, int ln16, int hi) {
  const unsigned short* p = w + (size_t)(m0 + ln16) * ldk + k0 + 8 * hi;
  BfTile u;
  u.q[0] = *(const uint4*)p;
  u.q[1] = *(const uint4*)(p + 16);
  return u.v;
}

// B tile 32x16 bf16 from [col][ldc] (channel-contiguous per column) storage.
// lane<16: col=col0+lane, K {k0..k0+7, k0+16..k0+23}; lanes>=16: +8 on both K chunks.
static __device__ inline v16bf load_b(const unsigned short* zl, int col0, int k0,
                                      int ldc, int ln16, int hi) {
  int c = col0 + ln16;
  if (c < 0) c = 0;                     // halo clamp (clamped cols are never consumed)
  const unsigned short* p = zl + (size_t)c * ldc + k0 + 8 * hi;
  BfTile u;
  u.q[0] = *(const uint4*)p;
  u.q[1] = *(const uint4*)(p + 16);
  return u.v;
}

// --------------------------- weight pre-conversion -------------------------
__global__ void convert_weights(const float* __restrict__ embed,
                                const float* __restrict__ wconv,
                                const float* __restrict__ wres,
                                const float* __restrict__ wskip,
                                const float* __restrict__ wout1,
                                const float* __restrict__ wout2,
                                unsigned short* __restrict__ ws) {
  int idx = blockIdx.x * 256 + threadIdx.x;
  if (idx >= WS_ELEMS) return;
  float v;
  if (idx < OFF_WCONV) {
    v = embed[idx];
  } else if (idx < OFF_WRES) {          // de-interleave conv taps: [i][o][c][k] -> [i][k][o][c]
    int d = idx - OFF_WCONV;
    int blk = d >> 14, rem = d & 16383;
    int tap = rem >> 13, oc = rem & 8191;
    int orow = oc >> 6, c = oc & 63;
    v = wconv[(size_t)blk * 16384 + orow * 128 + c * 2 + tap];
  } else if (idx < OFF_WSKIP) {
    v = wres[idx - OFF_WRES];
  } else if (idx < OFF_WOUT1) {
    v = wskip[idx - OFF_WSKIP];
  } else if (idx < OFF_WOUT2) {
    v = wout1[idx - OFF_WOUT1];
  } else {
    v = wout2[idx - OFF_WOUT2];
  }
  unsigned iv = __float_as_uint(v);
  ws[idx] = (unsigned short)((iv + 0x7FFFu + ((iv >> 16) & 1u)) >> 16);
}

// ------------------------------- fused kernel ------------------------------
__global__ __launch_bounds__(256) void wavenet_fused(
    const int* __restrict__ x, const float* __restrict__ h,
    const unsigned short* __restrict__ wsb, float* __restrict__ out) {
  // LDS: stage1: out stream [640][64] bf16 (80KB) + z staging [128][64] (16KB)
  //      stage2 (reuse): y [128][256] bf16 (64KB) + y1 [128][256] bf16 (64KB)
  __shared__ __align__(16) unsigned char smem[131072];
  unsigned short* out_lds = (unsigned short*)smem;                 // 0..80KB
  unsigned short* z_lds   = (unsigned short*)(smem + NC * 64 * 2); // 80..96KB
  unsigned short* y_lds   = (unsigned short*)smem;                 // 0..64KB
  unsigned short* y1_lds  = (unsigned short*)(smem + 65536);       // 64..128KB

  const int tid  = threadIdx.x;
  const int lane = tid & 31, wv = tid >> 5;      // 8 waves
  const int ln16 = lane & 15, hi = lane >> 4;
  const int tile = blockIdx.x & 63;
  const int b    = blockIdx.x >> 6;
  const int t0   = tile * TT;

  // ---- init residual stream with embedding gather (zeros where t<0: causal pad)
  for (int u = tid; u < NC * 8; u += 256) {
    int col = u >> 3, g = u & 7;
    int t = t0 - HALO + col;
    uint4 v = {0u, 0u, 0u, 0u};
    if (t >= 0) {
      int tok = x[b * Tlen + t];
      v = *(const uint4*)(wsb + OFF_EMB + tok * 64 + g * 8);
    }
    *(uint4*)(out_lds + (size_t)col * 64 + g * 8) = v;
  }
  __syncthreads();

  // skip accumulator: wave wv owns skip rows [32*wv, 32*wv+32) x all 128 cols
  v8f sk[2][8];
#pragma unroll
  for (int m = 0; m < 2; ++m)
#pragma unroll
    for (int n = 0; n < 8; ++n) sk[m][n] = vzero();

  for (int i = 0; i < NBk; ++i) {
    const int dil = kDil[i];
    const unsigned short* wcv = wsb + OFF_WCONV + (size_t)i * 16384; // [2][128][64]
    const unsigned short* wre = wsb + OFF_WRES  + (size_t)i * 4096;  // [64][64]
    const unsigned short* wsk = wsb + OFF_WSKIP + (size_t)i * 16384; // [256][64]
    const float* hb = h + ((size_t)b * 2048 + (size_t)i * 128) * Tlen;

    // right-to-left chunks so in-place residual update never clobbers left taps
    for (int c0 = HALO; c0 >= kCh0[i]; c0 -= TT) {
      // ---------------- phase A: wave wv -> N-tile at cols [c0+16wv, +16)
      const int colbase = c0 + wv * 16;
      const int t = t0 - HALO + colbase + ln16;
      if (c0 - TT >= kCh0[i] && t >= TT)
        __builtin_prefetch((const void*)(hb + (size_t)(8 * hi) * Tlen + (t - TT)), 0, 1);

      v8f g[8];                                  // 128x16 f32, init = h (conv bias term)
#pragma unroll
      for (int mt = 0; mt < 8; ++mt) {
        const float* hp = hb + (size_t)(mt * 16 + 8 * hi) * Tlen + t;
#pragma unroll
        for (int r = 0; r < 8; ++r) {
          float hv = 0.f;
          if (t >= 0) hv = hp[(size_t)r * Tlen];
          g[mt][r] = hv;
        }
      }
      v16bf b00 = load_b(out_lds, colbase - dil, 0,  64, ln16, hi);  // tap0 = x[t-dil]
      v16bf b01 = load_b(out_lds, colbase - dil, 32, 64, ln16, hi);
      v16bf b10 = load_b(out_lds, colbase,       0,  64, ln16, hi);  // tap1 = x[t]
      v16bf b11 = load_b(out_lds, colbase,       32, 64, ln16, hi);
#pragma unroll
      for (int mt = 0; mt < 8; ++mt) {
        v16bf a;
        a = load_a(wcv,        mt * 16, 0,  64, ln16, hi); g[mt] = wmma_bf(a, b00, g[mt]);
        a = load_a(wcv,        mt * 16, 32, 64, ln16, hi); g[mt] = wmma_bf(a, b01, g[mt]);
        a = load_a(wcv + 8192, mt * 16, 0,  64, ln16, hi); g[mt] = wmma_bf(a, b10, g[mt]);
        a = load_a(wcv + 8192, mt * 16, 32, 64, ln16, hi); g[mt] = wmma_bf(a, b11, g[mt]);
      }
      // gated activation: z = tanh(g[:64]) * sigmoid(g[64:]); D-layout -> z_lds B-layout
#pragma unroll
      for (int mt = 0; mt < 4; ++mt) {
        float zv[8];
#pragma unroll
        for (int r = 0; r < 8; ++r) {
          float tg = tanhf(g[mt][r]);
          float sg = 1.f / (1.f + __expf(-g[mt + 4][r]));
          zv[r] = tg * sg;
        }
        uint4 q;
        q.x = pk2(zv[0], zv[1]); q.y = pk2(zv[2], zv[3]);
        q.z = pk2(zv[4], zv[5]); q.w = pk2(zv[6], zv[7]);
        *(uint4*)(z_lds + (size_t)(wv * 16 + ln16) * 64 + mt * 16 + 8 * hi) = q;
      }
      __syncthreads();

      // ---------------- phase B: residual update + (final chunk) skip
      if (i < NBk - 1) {
        v16bf zb0 = load_b(z_lds, wv * 16, 0,  64, ln16, hi);
        v16bf zb1 = load_b(z_lds, wv * 16, 32, 64, ln16, hi);
        v8f rc[4];
#pragma unroll
        for (int mt = 0; mt < 4; ++mt) {
          v16bf a = load_a(wre, mt * 16, 0, 64, ln16, hi);
          rc[mt] = wmma_bf(a, zb0, vzero());
          a = load_a(wre, mt * 16, 32, 64, ln16, hi);
          rc[mt] = wmma_bf(a, zb1, rc[mt]);
        }
        unsigned short* op = out_lds + (size_t)(colbase + ln16) * 64 + 8 * hi;
#pragma unroll
        for (int mt = 0; mt < 4; ++mt) {             // out += Wres @ z (bf16 RMW)
          uint4 q = *(uint4*)(op + mt * 16);
          float f[8];
          f[0] = b2f(q.x & 0xFFFFu); f[1] = b2f(q.x >> 16);
          f[2] = b2f(q.y & 0xFFFFu); f[3] = b2f(q.y >> 16);
          f[4] = b2f(q.z & 0xFFFFu); f[5] = b2f(q.z >> 16);
          f[6] = b2f(q.w & 0xFFFFu); f[7] = b2f(q.w >> 16);
#pragma unroll
          for (int r = 0; r < 8; ++r) f[r] += rc[mt][r];
          q.x = pk2(f[0], f[1]); q.y = pk2(f[2], f[3]);
          q.z = pk2(f[4], f[5]); q.w = pk2(f[6], f[7]);
          *(uint4*)(op + mt * 16) = q;
        }
      }
      if (c0 == HALO) {                               // real output chunk: skip accum
#pragma unroll
        for (int nt = 0; nt < 8; ++nt) {
          v16bf zb0 = load_b(z_lds, nt * 16, 0,  64, ln16, hi);
          v16bf zb1 = load_b(z_lds, nt * 16, 32, 64, ln16, hi);
#pragma unroll
          for (int mt = 0; mt < 2; ++mt) {
            v16bf a = load_a(wsk, wv * 32 + mt * 16, 0, 64, ln16, hi);
            sk[mt][nt] = wmma_bf(a, zb0, sk[mt][nt]);
            a = load_a(wsk, wv * 32 + mt * 16, 32, 64, ln16, hi);
            sk[mt][nt] = wmma_bf(a, zb1, sk[mt][nt]);
          }
        }
      }
      __syncthreads();
    }
  }

  // ---------------- output head: y=relu(skip); y1=relu(W1@y); out=W2@y1
#pragma unroll
  for (int mt = 0; mt < 2; ++mt)
#pragma unroll
    for (int nt = 0; nt < 8; ++nt) {
      float f[8];
#pragma unroll
      for (int r = 0; r < 8; ++r) f[r] = fmaxf(sk[mt][nt][r], 0.f);
      uint4 q;
      q.x = pk2(f[0], f[1]); q.y = pk2(f[2], f[3]);
      q.z = pk2(f[4], f[5]); q.w = pk2(f[6], f[7]);
      *(uint4*)(y_lds + (size_t)(nt * 16 + ln16) * 256 + wv * 32 + mt * 16 + 8 * hi) = q;
    }
  __syncthreads();

  {
    v8f yc[16];
#pragma unroll
    for (int mt = 0; mt < 16; ++mt) yc[mt] = vzero();
#pragma unroll
    for (int kt = 0; kt < 8; ++kt) {
      v16bf bb = load_b(y_lds, wv * 16, kt * 32, 256, ln16, hi);
#pragma unroll
      for (int mt = 0; mt < 16; ++mt) {
        v16bf a = load_a(wsb + OFF_WOUT1, mt * 16, kt * 32, 256, ln16, hi);
        yc[mt] = wmma_bf(a, bb, yc[mt]);
      }
    }
#pragma unroll
    for (int mt = 0; mt < 16; ++mt) {
      float f[8];
#pragma unroll
      for (int r = 0; r < 8; ++r) f[r] = fmaxf(yc[mt][r], 0.f);
      uint4 q;
      q.x = pk2(f[0], f[1]); q.y = pk2(f[2], f[3]);
      q.z = pk2(f[4], f[5]); q.w = pk2(f[6], f[7]);
      *(uint4*)(y1_lds + (size_t)(wv * 16 + ln16) * 256 + mt * 16 + 8 * hi) = q;
    }
  }
  __syncthreads();

  {
    v8f oc[16];
#pragma unroll
    for (int mt = 0; mt < 16; ++mt) oc[mt] = vzero();
#pragma unroll
    for (int kt = 0; kt < 8; ++kt) {
      v16bf bb = load_b(y1_lds, wv * 16, kt * 32, 256, ln16, hi);
#pragma unroll
      for (int mt = 0; mt < 16; ++mt) {
        v16bf a = load_a(wsb + OFF_WOUT2, mt * 16, kt * 32, 256, ln16, hi);
        oc[mt] = wmma_bf(a, bb, oc[mt]);
      }
    }
    const int tcol = t0 + wv * 16 + ln16;
#pragma unroll
    for (int mt = 0; mt < 16; ++mt) {
      float* po = out + ((size_t)b * 256 + mt * 16 + 8 * hi) * Tlen + tcol;
#pragma unroll
      for (int r = 0; r < 8; ++r) po[(size_t)r * Tlen] = oc[mt][r];
    }
  }
}

// ------------------------------- launcher ----------------------------------
extern "C" void kernel_launch(void* const* d_in, const int* in_sizes, int n_in,
                              void* d_out, int out_size, void* d_ws, size_t ws_size,
                              hipStream_t stream) {
  (void)in_sizes; (void)n_in; (void)out_size; (void)ws_size;
  const int*   x     = (const int*)d_in[0];
  const float* h     = (const float*)d_in[1];
  const float* embed = (const float*)d_in[2];
  const float* wconv = (const float*)d_in[3];
  const float* wres  = (const float*)d_in[4];
  const float* wskip = (const float*)d_in[5];
  const float* wout1 = (const float*)d_in[6];
  const float* wout2 = (const float*)d_in[7];
  unsigned short* ws = (unsigned short*)d_ws;

  convert_weights<<<(WS_ELEMS + 255) / 256, 256, 0, stream>>>(
      embed, wconv, wres, wskip, wout1, wout2, ws);
  wavenet_fused<<<dim3(4 * (Tlen / TT)), dim3(256), 0, stream>>>(
      x, h, ws, (float*)d_out);
}
